// RKANSubLayer_35914516529363
// MI455X (gfx1250) — matrix operand order repaired and checked by reading, not compiled, not run on hardware
//
#include <hip/hip_runtime.h>
#include <hip/hip_bf16.h>
#include <math.h>

// ---------------------------------------------------------------------------
// Types for CDNA5 WMMA (wave32): v_wmma_f32_16x16x32_bf16
// ---------------------------------------------------------------------------
typedef __attribute__((ext_vector_type(16))) __bf16 v16bf;
typedef __attribute__((ext_vector_type(8)))  __bf16 v8bf;   // 16 bytes
typedef __attribute__((ext_vector_type(8)))  float  v8f;

#define B_ROWS 4096
#define IN_DIM 1024
#define H_DIM  1024
#define NCOEF  8
#define KCAT   (IN_DIM * (1 + NCOEF))   // 9216 = [silu | 8 spline bases] per input feature

// ---------------------------------------------------------------------------
// CDNA5 async global->LDS copy (ASYNCcnt-tracked, bypasses VGPR staging).
// INST_OFFSET applies to BOTH the global and the LDS address (ISA 15.18.3),
// so one (lds_off, gaddr) pair covers 32 contiguous bytes with offset:16.
// lds_off is the wave-relative LDS byte offset = low 32 bits of the generic
// pointer to a __shared__ object.
// ---------------------------------------------------------------------------
__device__ __forceinline__ void async_copy_32B(unsigned lds_off, const void* gptr) {
    asm volatile(
        "global_load_async_to_lds_b128 %0, %1, off\n\t"
        "global_load_async_to_lds_b128 %0, %1, off offset:16"
        :: "v"(lds_off), "v"(gptr) : "memory");
}

__device__ __forceinline__ void wait_async0() {
#if __has_builtin(__builtin_amdgcn_s_wait_asynccnt)
    __builtin_amdgcn_s_wait_asynccnt(0);
#else
    asm volatile("s_wait_asynccnt 0x0" ::: "memory");
#endif
}

// ---------------------------------------------------------------------------
// Cubic B-spline bases (Cox-de Boor), grid_size=5, k=3, uniform knots on [-1,1]
// knots t[j] = -1 + (j-3)*0.4, j = 0..11 ; produces 8 bases
// ---------------------------------------------------------------------------
__device__ __forceinline__ void bspline8(float x, float* __restrict__ out) {
    const float h = 0.4f;
    float t[12];
#pragma unroll
    for (int j = 0; j < 12; ++j) t[j] = -1.0f + (float)(j - 3) * h;
    float b[11];
#pragma unroll
    for (int j = 0; j < 11; ++j) b[j] = (x >= t[j] && x < t[j + 1]) ? 1.0f : 0.0f;
#pragma unroll
    for (int d = 1; d <= 3; ++d) {
#pragma unroll
        for (int j = 0; j < 11 - 1; ++j) {
            if (j < 11 - d) {
                float left  = (x - t[j]) / (t[j + d] - t[j]) * b[j];
                float right = (t[j + d + 1] - x) / (t[j + d + 1] - t[j + 1]) * b[j + 1];
                b[j] = left + right;
            }
        }
    }
#pragma unroll
    for (int c = 0; c < NCOEF; ++c) out[c] = b[c];
}

// ---------------------------------------------------------------------------
// prep_act: act (B x IN f32) -> Acat (B x 9216 bf16) = [silu(act) | bsp(act)]
// ---------------------------------------------------------------------------
__global__ void prep_act_kernel(const float* __restrict__ act, __bf16* __restrict__ A, int n) {
    int idx = blockIdx.x * blockDim.x + threadIdx.x;
    if (idx >= n) return;
    int b = idx / IN_DIM, i = idx % IN_DIM;
    float x = act[idx];
    float s = x / (1.0f + __expf(-x));          // silu
    __bf16* row = A + (size_t)b * KCAT;
    row[i] = (__bf16)s;
    float bs[NCOEF];
    bspline8(x, bs);
    v8bf pk;
#pragma unroll
    for (int c = 0; c < NCOEF; ++c) pk[c] = (__bf16)bs[c];
    *(v8bf*)(row + IN_DIM + (size_t)i * NCOEF) = pk;   // 16B aligned: 2048 + 16*i bytes
}

// ---------------------------------------------------------------------------
// prep_w: (base_w, spline_w, scaler) -> Wt (OUT x 9216 bf16), K-major,
// Wt[n][k] matches Acat's k ordering (so GEMM computes x @ W^T exactly).
// ---------------------------------------------------------------------------
__global__ void prep_w_kernel(const float* __restrict__ base_w,
                              const float* __restrict__ spline_w,
                              const float* __restrict__ scaler,
                              __bf16* __restrict__ W, int n /* = OUT*IN */) {
    int idx = blockIdx.x * blockDim.x + threadIdx.x;
    if (idx >= n) return;
    int o = idx / IN_DIM, i = idx % IN_DIM;
    __bf16* row = W + (size_t)o * KCAT;
    row[i] = (__bf16)base_w[idx];
    float sc = scaler[idx];
    const float* sp = spline_w + (size_t)idx * NCOEF;
    v8bf pk;
#pragma unroll
    for (int c = 0; c < NCOEF; ++c) pk[c] = (__bf16)(sp[c] * sc);
    *(v8bf*)(row + IN_DIM + (size_t)i * NCOEF) = pk;
}

// ---------------------------------------------------------------------------
// misc elementwise kernels
// ---------------------------------------------------------------------------
__global__ void f32_to_bf16_kernel(const float* __restrict__ in, __bf16* __restrict__ out, int n) {
    int i = blockIdx.x * blockDim.x + threadIdx.x;
    if (i < n) out[i] = (__bf16)in[i];
}
__global__ void fill_zero_kernel(float* __restrict__ p, int n) {
    int i = blockIdx.x * blockDim.x + threadIdx.x;
    if (i < n) p[i] = 0.0f;
}
__global__ void fill_bias_kernel(float* __restrict__ p, const float* __restrict__ bias, int n) {
    int i = blockIdx.x * blockDim.x + threadIdx.x;
    if (i < n) p[i] = bias[i & (H_DIM - 1)];
}
__global__ void tanh_kernel(const float* __restrict__ s, float* __restrict__ o, int n) {
    int i = blockIdx.x * blockDim.x + threadIdx.x;
    if (i < n) o[i] = tanhf(s[i]);
}
__global__ void copy_kernel(const float* __restrict__ in, float* __restrict__ out, int n) {
    int i = blockIdx.x * blockDim.x + threadIdx.x;
    if (i < n) out[i] = in[i];
}

// ---------------------------------------------------------------------------
// bf16 WMMA GEMM:  C(M x N, f32) += A(M x K, bf16, K-major) * Wt(N x K, bf16, K-major)^T
// 256 threads = 8 waves; block tile 128x128; K-step 32.
// Double-buffered LDS; tiles staged with GLOBAL_LOAD_ASYNC_TO_LDS_B128 so the
// DMA for K-step k+1 overlaps the 8 WMMAs of K-step k; one barrier + one
// s_wait_asynccnt per iteration.
// Fragment layouts per CDNA5 ISA 7.12.2:
//   A 16x32 bf16: lane<16 -> row=lane,   K {0..7,16..23}; lane>=16 -> row=lane-16, K {8..15,24..31}
//   B 32x16 bf16: lane<16 -> col=lane,   K {0..7,16..23}; lane>=16 -> col=lane-16, K {8..15,24..31}
//   C 16x16 f32 : VGPR j  -> M = j (lanes 0..15) / j+8 (lanes 16..31), N = lane%16
// ---------------------------------------------------------------------------
#define TM 128
#define TN 128
#define TK 32
#define LPAD 8   // pad LDS rows to 40 bf16 = 80B (multiple of 16B, spreads banks)

__global__ __launch_bounds__(256)
void gemm_bf16_kernel(const __bf16* __restrict__ A, const __bf16* __restrict__ Wt,
                      float* __restrict__ C, int M, int N, int K) {
    __shared__ __bf16 As[2][TM][TK + LPAD];
    __shared__ __bf16 Bs[2][TN][TK + LPAD];

    const int tid   = threadIdx.x;
    const int lane  = tid & 31;
    const int wave  = tid >> 5;
    const int waveM = wave & 3;        // 0..3
    const int waveN = wave >> 2;       // 0..1
    const int bm    = blockIdx.y * TM;
    const int bn    = blockIdx.x * TN;

    v8f acc[2][4];
#pragma unroll
    for (int a = 0; a < 2; ++a)
#pragma unroll
        for (int b = 0; b < 4; ++b)
#pragma unroll
            for (int j = 0; j < 8; ++j) acc[a][b][j] = 0.0f;

    // global->LDS mapping: each thread async-copies 32B of A-tile and 32B of B-tile
    const int lrow  = tid >> 1;            // 0..127
    const int lhalf = (tid & 1) * 16;      // 0 or 16

    const unsigned ldsA[2] = { (unsigned)(size_t)&As[0][lrow][lhalf],
                               (unsigned)(size_t)&As[1][lrow][lhalf] };
    const unsigned ldsB[2] = { (unsigned)(size_t)&Bs[0][lrow][lhalf],
                               (unsigned)(size_t)&Bs[1][lrow][lhalf] };
    const __bf16* gA = A  + (size_t)(bm + lrow) * K + lhalf;
    const __bf16* gB = Wt + (size_t)(bn + lrow) * K + lhalf;

    const int halfsel = lane >> 4;         // 0 | 1
    const int lr      = lane & 15;
    const int ka0     = halfsel * 8;       // A frag chunk0 K-offset (0 | 8)
    const int ka1     = ka0 + 16;          // A frag chunk1 K-offset (16 | 24)
    const int kb      = halfsel * 16;      // unused (0 | 16); B frag uses ka0/ka1

    // prologue: stage K-step 0 into buffer 0
    async_copy_32B(ldsA[0], gA);
    async_copy_32B(ldsB[0], gB);

    int buf = 0;
    for (int k0 = 0; k0 < K; k0 += TK) {
        wait_async0();        // this wave's stage of `buf` is in LDS
        __syncthreads();      // everyone's stage is in LDS; prior reads of buf^1 done

        if (k0 + TK < K) {    // prefetch next K-step into the other buffer (overlaps WMMAs)
            async_copy_32B(ldsA[buf ^ 1], gA + (k0 + TK));
            async_copy_32B(ldsB[buf ^ 1], gB + (k0 + TK));
        }

        v16bf afrag[2];
#pragma unroll
        for (int am = 0; am < 2; ++am) {
            int row = waveM * 32 + am * 16 + lr;
            v8bf lo = *(const v8bf*)&As[buf][row][ka0];
            v8bf hi = *(const v8bf*)&As[buf][row][ka1];
            afrag[am] = __builtin_shufflevector(lo, hi, 0,1,2,3,4,5,6,7,8,9,10,11,12,13,14,15);
        }
        v16bf bfrag[4];
#pragma unroll
        for (int bv = 0; bv < 4; ++bv) {
            int col = waveN * 64 + bv * 16 + lr;
            v8bf lo = *(const v8bf*)&Bs[buf][col][ka0];
            v8bf hi = *(const v8bf*)&Bs[buf][col][ka1];
            bfrag[bv] = __builtin_shufflevector(lo, hi, 0,1,2,3,4,5,6,7,8,9,10,11,12,13,14,15);
        }

#pragma unroll
        for (int am = 0; am < 2; ++am)
#pragma unroll
            for (int bv = 0; bv < 4; ++bv)
                acc[am][bv] = __builtin_amdgcn_wmma_f32_16x16x32_bf16(
                    /*neg_a=*/false, afrag[am],
                    /*neg_b=*/false, bfrag[bv],
                    /*c_mod=*/(short)0, acc[am][bv],
                    /*reuse_a=*/false, /*reuse_b=*/false);

        buf ^= 1;
    }

    // C += acc  (each element owned by exactly one lane; kernels run sequentially)
    const int rofs = halfsel * 8;
#pragma unroll
    for (int am = 0; am < 2; ++am) {
#pragma unroll
        for (int bv = 0; bv < 4; ++bv) {
            int col = bn + waveN * 64 + bv * 16 + lr;
#pragma unroll
            for (int j = 0; j < 8; ++j) {
                int row = bm + waveM * 32 + am * 16 + rofs + j;
                C[(size_t)row * N + col] += acc[am][bv][j];
            }
        }
    }
}

// ---------------------------------------------------------------------------
// kernel_launch
// inputs: 0 x_t, 1 h_prev(unused), 2 z_prev, 3 wx_base, 4 wx_spline, 5 wx_scaler,
//         6 wh_base, 7 wh_spline, 8 wh_scaler, 9 hz_base, 10 hz_spline, 11 hz_scaler,
//         12 hh_w, 13 hh_b
// d_out: [o_t (4096x1024 f32) | z_t (4096x1024 f32)]
// ---------------------------------------------------------------------------
extern "C" void kernel_launch(void* const* d_in, const int* in_sizes, int n_in,
                              void* d_out, int out_size, void* d_ws, size_t ws_size,
                              hipStream_t stream) {
    (void)in_sizes; (void)n_in; (void)out_size; (void)ws_size;

    const float* x_t       = (const float*)d_in[0];
    const float* z_prev    = (const float*)d_in[2];
    const float* wx_base   = (const float*)d_in[3];
    const float* wx_spline = (const float*)d_in[4];
    const float* wx_scaler = (const float*)d_in[5];
    const float* wh_base   = (const float*)d_in[6];
    const float* wh_spline = (const float*)d_in[7];
    const float* wh_scaler = (const float*)d_in[8];
    const float* hz_base   = (const float*)d_in[9];
    const float* hz_spline = (const float*)d_in[10];
    const float* hz_scaler = (const float*)d_in[11];
    const float* hh_w      = (const float*)d_in[12];
    const float* hh_b      = (const float*)d_in[13];

    float* o_out = (float*)d_out;                                  // o_t
    float* z_out = (float*)d_out + (size_t)B_ROWS * H_DIM;         // z_t

    // workspace carve-out (buffers reused across the three KAN layers)
    size_t off = 0;
    auto carve = [&](size_t bytes) {
        void* p = (char*)d_ws + off;
        off += (bytes + 255) & ~(size_t)255;
        return p;
    };
    __bf16* Acat = (__bf16*)carve((size_t)B_ROWS * KCAT * sizeof(__bf16));   // 75.5 MB
    __bf16* Wcat = (__bf16*)carve((size_t)H_DIM * KCAT * sizeof(__bf16));    // 18.9 MB
    float*  S    = (float*) carve((size_t)B_ROWS * H_DIM * sizeof(float));   // 16 MB
    float*  Z    = (float*) carve((size_t)B_ROWS * H_DIM * sizeof(float));   // 16 MB
    __bf16* zbf  = (__bf16*)carve((size_t)B_ROWS * H_DIM * sizeof(__bf16));  // 8 MB
    __bf16* hhbf = (__bf16*)carve((size_t)H_DIM * H_DIM * sizeof(__bf16));   // 2 MB

    const int nBH = B_ROWS * H_DIM;     // 4M
    const int nW  = H_DIM * IN_DIM;     // 1M
    const int TPB = 256;
    dim3 ggrid(H_DIM / TN, B_ROWS / TM);  // (8, 32)

    // S = 0 ; Z = bias
    fill_zero_kernel<<<(nBH + TPB - 1) / TPB, TPB, 0, stream>>>(S, nBH);
    fill_bias_kernel<<<(nBH + TPB - 1) / TPB, TPB, 0, stream>>>(Z, hh_b, nBH);

    // S += kan_linear(x_t; wx)
    prep_act_kernel<<<(nBH + TPB - 1) / TPB, TPB, 0, stream>>>(x_t, Acat, nBH);
    prep_w_kernel<<<(nW + TPB - 1) / TPB, TPB, 0, stream>>>(wx_base, wx_spline, wx_scaler, Wcat, nW);
    gemm_bf16_kernel<<<ggrid, TPB, 0, stream>>>(Acat, Wcat, S, B_ROWS, H_DIM, KCAT);

    // S += kan_linear(z_prev; wh)
    prep_act_kernel<<<(nBH + TPB - 1) / TPB, TPB, 0, stream>>>(z_prev, Acat, nBH);
    prep_w_kernel<<<(nW + TPB - 1) / TPB, TPB, 0, stream>>>(wh_base, wh_spline, wh_scaler, Wcat, nW);
    gemm_bf16_kernel<<<ggrid, TPB, 0, stream>>>(Acat, Wcat, S, B_ROWS, H_DIM, KCAT);

    // o_t = tanh(S)
    tanh_kernel<<<(nBH + TPB - 1) / TPB, TPB, 0, stream>>>(S, o_out, nBH);

    // Z += z_prev @ hh_w^T   (K = 1024, plain bf16 GEMM)
    f32_to_bf16_kernel<<<(nBH + TPB - 1) / TPB, TPB, 0, stream>>>(z_prev, zbf, nBH);
    f32_to_bf16_kernel<<<(nW + TPB - 1) / TPB, TPB, 0, stream>>>(hh_w, hhbf, nW);
    gemm_bf16_kernel<<<ggrid, TPB, 0, stream>>>(zbf, hhbf, Z, B_ROWS, H_DIM, H_DIM);

    // Z += kan_linear(o_t; hz)
    prep_act_kernel<<<(nBH + TPB - 1) / TPB, TPB, 0, stream>>>(o_out, Acat, nBH);
    prep_w_kernel<<<(nW + TPB - 1) / TPB, TPB, 0, stream>>>(hz_base, hz_spline, hz_scaler, Wcat, nW);
    gemm_bf16_kernel<<<ggrid, TPB, 0, stream>>>(Acat, Wcat, Z, B_ROWS, H_DIM, KCAT);

    // z_t out
    copy_kernel<<<(nBH + TPB - 1) / TPB, TPB, 0, stream>>>(Z, z_out, nBH);
}
